// STAN_11605001634102
// MI455X (gfx1250) — matrix operand order repaired and checked by reading, not compiled, not run on hardware
//
#include <hip/hip_runtime.h>
#include <hip/hip_bf16.h>
#include <math.h>

// ---------------------------------------------------------------- constants
#define B_  8
#define N_  2000
#define T_  8
#define F_  128
#define DEG_ 8
#define H_  4
#define D1_ 64
#define D2_ 64
#define G_  64
#define PW_ 16
#define BT_ (B_*T_)          // 64
#define NT_ (N_*T_)          // 16000
#define M_GI (NT_*B_)        // 128000 rows for gi GEMM
#define NTILE_ ((N_ + 15) / 16)   // 125

typedef __bf16 bf16_t;
typedef bf16_t v16bf __attribute__((ext_vector_type(16)));
typedef float  v8f   __attribute__((ext_vector_type(8)));
typedef unsigned int u32x4 __attribute__((ext_vector_type(4)));
typedef int          i32x8 __attribute__((ext_vector_type(8)));
typedef int          i32x4 __attribute__((ext_vector_type(4)));

// hardware round-to-nearest-even convert (v_cvt on gfx1250)
__device__ __forceinline__ bf16_t f2bf(float f) { return (bf16_t)f; }

// A-fragment element e (0..15) of lane (hi = lane>>4) maps to K offset
__device__ __forceinline__ int a_koff(int e, int hi) {
    return ((e & 8) << 1) + hi * 8 + (e & 7);   // {0..7,16..23} + hi*8
}

__device__ __forceinline__ float lrelu(float x) { return x > 0.f ? x : 0.01f * x; }
__device__ __forceinline__ float elu(float x)   { return x > 0.f ? x : (expf(x) - 1.f); }
__device__ __forceinline__ float sigm(float x)  { return 1.f / (1.f + expf(-x)); }

// -------- Tensor Data Mover: 2-D tile (bf16) global -> LDS  ---------------
// D# per CDNA5 ISA ch.8: group0 = {count, lds_addr, global_addr, type=2},
// group1 = {data_size=2B, tensor dims, tile dims, stride}; single-tile load.
// This toolchain exposes the 6-arg builtin (clang-23 / therock-10.0 form):
//   (u32x4 g0, i32x8 g1, i32x4 g2, i32x4 g3, i32x8 g4, i32 cpol)
#if __has_builtin(__builtin_amdgcn_tensor_load_to_lds)
#define HAVE_TDM 1
__device__ __forceinline__ void tdm_load_2d_bf16(unsigned lds_off, const void* g,
                                                 unsigned d0, unsigned d1,
                                                 unsigned stride0) {
    unsigned long long ga = (unsigned long long)g;
    u32x4 g0;
    g0[0] = 1u;                                         // count=1, user mode
    g0[1] = lds_off;                                    // lds_addr
    g0[2] = (unsigned)ga;                               // global_addr[31:0]
    g0[3] = (unsigned)((ga >> 32) & 0x01FFFFFFu) | (2u << 30);  // addr hi | type=2
    i32x8 g1;
    g1[0] = (int)(1u << 16);                            // data_size = 2 bytes
    g1[1] = (int)((d0 & 0xFFFFu) << 16);                // tensor_dim0 lo16
    g1[2] = (int)(((d0 >> 16) & 0xFFFFu) | ((d1 & 0xFFFFu) << 16)); // d0 hi | d1 lo
    g1[3] = (int)(((d1 >> 16) & 0xFFFFu) | ((d0 & 0xFFFFu) << 16)); // d1 hi | tile_dim0
    g1[4] = (int)(d1 & 0xFFFFu);                        // tile_dim1 (tile_dim2=0)
    g1[5] = (int)stride0;                               // tensor_dim0_stride lo32
    g1[6] = 0; g1[7] = 0;
    i32x4 z4 = {0, 0, 0, 0};
    i32x8 z8 = {0, 0, 0, 0, 0, 0, 0, 0};
    __builtin_amdgcn_tensor_load_to_lds(g0, g1, z4, z4, z8, 0);
}
#else
#define HAVE_TDM 0
#endif

// ---------------------------------------------------------------- kernel 0
// pack W1 (32768) -> bf16, W2 (16384) -> bf16, Wih (192x64) -> WihT (64x192)
__global__ void k_pack_weights(const float* __restrict__ W1,
                               const float* __restrict__ W2,
                               const float* __restrict__ Wih,
                               bf16_t* __restrict__ W1b,
                               bf16_t* __restrict__ W2b,
                               bf16_t* __restrict__ WihTb) {
    int i = blockIdx.x * blockDim.x + threadIdx.x;
    if (i < 32768)                  W1b[i] = f2bf(W1[i]);
    else if (i < 32768 + 16384)     W2b[i - 32768] = f2bf(W2[i - 32768]);
    else if (i < 32768 + 16384 + 12288) {
        int j = i - 49152;                  // Wih index: row j/64 (0..191), k j%64
        int row = j >> 6, k = j & 63;
        WihTb[k * 192 + row] = f2bf(Wih[j]);
    }
}

// ---------------------------------------------------------------- kernel 1
// GAT1 linear: z1[bt,h,n,:] = x @ W1[h] + b1[h]; attention dots fused.
// 4 waves/block, W1 head tile (128x64 bf16, 16 KB) TDM-staged in LDS.
__global__ __launch_bounds__(128)
void k_gat1_gemm(const float* __restrict__ dynamic,      // (B,N,T,F)
                 const bf16_t* __restrict__ W1b,         // (H,F,D1)
                 const float* __restrict__ b1,
                 const float* __restrict__ a1,
                 float* __restrict__ z1,                 // (BT,H,N,D1)
                 float* __restrict__ s1s,
                 float* __restrict__ s1d) {
    __shared__ __align__(16) unsigned char smem[16384 + 4 * 16 * 64 * 4]; // W + zs
    bf16_t* sW = (bf16_t*)smem;                          // LDS offset 0
    float*  zs = (float*)(smem + 16384);

    const int tid = threadIdx.x, lane = tid & 31, wave = tid >> 5;
    const int m = lane & 15, hi = lane >> 4;
    const int h = blockIdx.y, bt = blockIdx.z;
    const int b = bt >> 3, t = bt & 7;
    const int tile = blockIdx.x * 4 + wave;
    const bool valid = tile < NTILE_;
    const int n0 = (valid ? tile : 0) * 16;

#if HAVE_TDM
    if (wave == 0) {
        tdm_load_2d_bf16(0u, W1b + (size_t)h * F_ * D1_, 64u, 128u, 64u);
        __builtin_amdgcn_s_wait_tensorcnt(0);
    }
#else
    for (int i = tid; i < F_ * D1_; i += 128) sW[i] = W1b[(size_t)h * F_ * D1_ + i];
#endif
    __syncthreads();

    v8f acc[4] = { {}, {}, {}, {} };
    const float* pA = dynamic + (((size_t)(b * N_ + (n0 + m)) * T_) + t) * F_;

    #pragma unroll
    for (int kc = 0; kc < 4; ++kc) {                    // K = 128
        v16bf a;
        #pragma unroll
        for (int e = 0; e < 16; ++e)
            a[e] = f2bf(pA[kc * 32 + a_koff(e, hi)]);
        #pragma unroll
        for (int tn = 0; tn < 4; ++tn) {
            v16bf bb;
            const bf16_t* pBk = sW + (size_t)(kc * 32) * 64 + tn * 16 + m;
            #pragma unroll
            for (int e = 0; e < 16; ++e) bb[e] = pBk[a_koff(e, hi) * 64];
            acc[tn] = __builtin_amdgcn_wmma_f32_16x16x32_bf16(
                false, a, false, bb, (short)0, acc[tn], false, false);
        }
    }
    float* wz = zs + wave * 16 * 64;
    #pragma unroll
    for (int tn = 0; tn < 4; ++tn)
        #pragma unroll
        for (int r = 0; r < 8; ++r)
            wz[(r + hi * 8) * 64 + tn * 16 + m] = acc[tn][r];
    __syncthreads();

    if (lane < 16) {                                    // bias + attention dots
        float ss = 0.f, sd = 0.f;
        #pragma unroll 8
        for (int d = 0; d < 64; ++d) {
            float zb = wz[lane * 64 + d] + b1[h * 64 + d];
            wz[lane * 64 + d] = zb;
            ss += zb * a1[h * 128 + d];
            sd += zb * a1[h * 128 + 64 + d];
        }
        if (valid) {
            size_t sidx = ((size_t)(bt * H_ + h)) * N_ + (n0 + lane);
            s1s[sidx] = ss;
            s1d[sidx] = sd;
        }
    }
    __syncthreads();
    if (valid) {
        float4* dst = (float4*)(z1 + (((size_t)(bt * H_ + h)) * N_ + n0) * 64);
        const float4* srcv = (const float4*)wz;
        #pragma unroll
        for (int i = 0; i < 8; ++i) dst[i * 32 + lane] = srcv[i * 32 + lane];
    }
}

// ---------------------------------------------------------------- kernel 2
// GAT1 attention softmax over DEG neighbors + aggregate + ELU -> h1 (bf16)
__global__ __launch_bounds__(256)
void k_gat1_aggr(const int* __restrict__ src,
                 const float* __restrict__ z1,
                 const float* __restrict__ s1s,
                 const float* __restrict__ s1d,
                 const float* __restrict__ a1b,
                 bf16_t* __restrict__ h1) {              // (BT,N,H*D1)
    const int tid = threadIdx.x;
    const int h = tid >> 6, d = tid & 63;
    const int n = blockIdx.x, bt = blockIdx.y;
    const size_t sbase = ((size_t)(bt * H_ + h)) * N_;
    const size_t zbase = sbase * 64;
    const float sdn = s1d[sbase + n] + a1b[h];

    int   nb[DEG_];
    float sc[DEG_], mx = -1e30f;
    #pragma unroll
    for (int k = 0; k < DEG_; ++k) {
        int s = src[n * DEG_ + k];
        nb[k] = s;
        float e = lrelu(s1s[sbase + s] + sdn);
        sc[k] = e;
        mx = fmaxf(mx, e);
    }
    float sum = 0.f;
    #pragma unroll
    for (int k = 0; k < DEG_; ++k) { sc[k] = expf(sc[k] - mx); sum += sc[k]; }
    float inv = 1.f / sum, accv = 0.f;
    #pragma unroll
    for (int k = 0; k < DEG_; ++k)
        accv += sc[k] * inv * z1[zbase + (size_t)nb[k] * 64 + d];
    h1[((size_t)bt * N_ + n) * 256 + h * 64 + d] = f2bf(elu(accv));
}

// ---------------------------------------------------------------- kernel 3
// GAT2 linear: z2 = h1 @ W2 + b2 (K=256). W2 (256x64, 32 KB) TDM-staged.
__global__ __launch_bounds__(128)
void k_gat2_gemm(const bf16_t* __restrict__ h1,
                 const bf16_t* __restrict__ W2b,
                 const float* __restrict__ b2,
                 const float* __restrict__ a2,
                 float* __restrict__ z2,                 // (BT,N,64)
                 float* __restrict__ s2s,
                 float* __restrict__ s2d) {
    __shared__ __align__(16) unsigned char smem[32768 + 4 * 16 * 64 * 4];
    bf16_t* sW = (bf16_t*)smem;                          // LDS offset 0
    float*  zs = (float*)(smem + 32768);

    const int tid = threadIdx.x, lane = tid & 31, wave = tid >> 5;
    const int m = lane & 15, hi = lane >> 4;
    const int bt = blockIdx.y;
    const int tile = blockIdx.x * 4 + wave;
    const bool valid = tile < NTILE_;
    const int n0 = (valid ? tile : 0) * 16;

#if HAVE_TDM
    if (wave == 0) {
        tdm_load_2d_bf16(0u, W2b, 64u, 256u, 64u);
        __builtin_amdgcn_s_wait_tensorcnt(0);
    }
#else
    for (int i = tid; i < 256 * 64; i += 128) sW[i] = W2b[i];
#endif
    __syncthreads();

    v8f acc[4] = { {}, {}, {}, {} };
    const bf16_t* pA = h1 + ((size_t)bt * N_ + (n0 + m)) * 256;

    #pragma unroll
    for (int kc = 0; kc < 8; ++kc) {                    // K = 256
        v16bf a;
        #pragma unroll
        for (int e = 0; e < 16; ++e) a[e] = pA[kc * 32 + a_koff(e, hi)];
        #pragma unroll
        for (int tn = 0; tn < 4; ++tn) {
            v16bf bb;
            const bf16_t* pBk = sW + (size_t)(kc * 32) * 64 + tn * 16 + m;
            #pragma unroll
            for (int e = 0; e < 16; ++e) bb[e] = pBk[a_koff(e, hi) * 64];
            acc[tn] = __builtin_amdgcn_wmma_f32_16x16x32_bf16(
                false, a, false, bb, (short)0, acc[tn], false, false);
        }
    }
    float* wz = zs + wave * 16 * 64;
    #pragma unroll
    for (int tn = 0; tn < 4; ++tn)
        #pragma unroll
        for (int r = 0; r < 8; ++r)
            wz[(r + hi * 8) * 64 + tn * 16 + m] = acc[tn][r];
    __syncthreads();

    if (lane < 16) {
        float ss = 0.f, sd = 0.f;
        #pragma unroll 8
        for (int d = 0; d < 64; ++d) {
            float zb = wz[lane * 64 + d] + b2[d];
            wz[lane * 64 + d] = zb;
            ss += zb * a2[d];
            sd += zb * a2[64 + d];
        }
        if (valid) {
            size_t sidx = (size_t)bt * N_ + (n0 + lane);
            s2s[sidx] = ss;
            s2d[sidx] = sd;
        }
    }
    __syncthreads();
    if (valid) {
        float4* dst = (float4*)(z2 + ((size_t)bt * N_ + n0) * 64);
        const float4* srcv = (const float4*)wz;
        #pragma unroll
        for (int i = 0; i < 8; ++i) dst[i * 32 + lane] = srcv[i * 32 + lane];
    }
}

// ---------------------------------------------------------------- kernel 4
// GAT2 aggregate + ELU -> seq in GRU layout ((n*T+t)*B + b, 64), bf16
__global__ __launch_bounds__(64)
void k_gat2_aggr(const int* __restrict__ src,
                 const float* __restrict__ z2,
                 const float* __restrict__ s2s,
                 const float* __restrict__ s2d,
                 const float* __restrict__ a2b,
                 bf16_t* __restrict__ seq) {
    const int d = threadIdx.x;
    const int n = blockIdx.x, bt = blockIdx.y;
    const int b = bt >> 3, t = bt & 7;
    const size_t sbase = (size_t)bt * N_;
    const size_t zbase = sbase * 64;
    const float sdn = s2d[sbase + n] + a2b[0];

    int   nb[DEG_];
    float sc[DEG_], mx = -1e30f;
    #pragma unroll
    for (int k = 0; k < DEG_; ++k) {
        int s = src[n * DEG_ + k];
        nb[k] = s;
        float e = lrelu(s2s[sbase + s] + sdn);
        sc[k] = e;
        mx = fmaxf(mx, e);
    }
    float sum = 0.f;
    #pragma unroll
    for (int k = 0; k < DEG_; ++k) { sc[k] = expf(sc[k] - mx); sum += sc[k]; }
    float inv = 1.f / sum, accv = 0.f;
    #pragma unroll
    for (int k = 0; k < DEG_; ++k)
        accv += sc[k] * inv * z2[zbase + (size_t)nb[k] * 64 + d];
    seq[((size_t)(n * T_ + t) * B_ + b) * 64 + d] = f2bf(elu(accv));
}

// ---------------------------------------------------------------- kernel 5
// gi = seq @ WihT + bih : (128000x64)@(64x192), bf16 out. WihT TDM-staged.
__global__ __launch_bounds__(128)
void k_gi_gemm(const bf16_t* __restrict__ seq,
               const bf16_t* __restrict__ WihTb,         // (64,192) row-major
               const float* __restrict__ bih,
               bf16_t* __restrict__ gi) {                // (128000,192) bf16
    __shared__ __align__(16) unsigned char smem[24576 + 4 * 16 * 192 * 2];
    bf16_t* sW = (bf16_t*)smem;                          // LDS offset 0
    bf16_t* gz = (bf16_t*)(smem + 24576);

    const int tid = threadIdx.x, lane = tid & 31, wave = tid >> 5;
    const int m = lane & 15, hi = lane >> 4;
    const int m0 = (blockIdx.x * 4 + wave) * 16;

#if HAVE_TDM
    if (wave == 0) {
        tdm_load_2d_bf16(0u, WihTb, 192u, 64u, 192u);
        __builtin_amdgcn_s_wait_tensorcnt(0);
    }
#else
    for (int i = tid; i < 64 * 192; i += 128) sW[i] = WihTb[i];
#endif
    __syncthreads();

    v8f acc[12];
    #pragma unroll
    for (int j = 0; j < 12; ++j) acc[j] = (v8f){};

    const bf16_t* pA = seq + (size_t)(m0 + m) * 64;
    #pragma unroll
    for (int kc = 0; kc < 2; ++kc) {                    // K = 64
        v16bf a;
        #pragma unroll
        for (int e = 0; e < 16; ++e) a[e] = pA[kc * 32 + a_koff(e, hi)];
        #pragma unroll
        for (int jt = 0; jt < 12; ++jt) {
            v16bf bb;
            const bf16_t* pBk = sW + (size_t)(kc * 32) * 192 + jt * 16 + m;
            #pragma unroll
            for (int e = 0; e < 16; ++e) bb[e] = pBk[a_koff(e, hi) * 192];
            acc[jt] = __builtin_amdgcn_wmma_f32_16x16x32_bf16(
                false, a, false, bb, (short)0, acc[jt], false, false);
        }
    }
    bf16_t* wz = gz + wave * 16 * 192;
    #pragma unroll
    for (int jt = 0; jt < 12; ++jt)
        #pragma unroll
        for (int r = 0; r < 8; ++r)
            wz[(r + hi * 8) * 192 + jt * 16 + m] = f2bf(acc[jt][r]);
    __syncthreads();
    bf16_t* dst = gi + (size_t)m0 * 192;
    for (int i = lane; i < 16 * 192; i += 32)
        dst[i] = f2bf((float)wz[i] + bih[i % 192]);
}

// ---------------------------------------------------------------- kernel 6
// persistent sequential GRU over 16000 steps + fused projection
__global__ __launch_bounds__(512)
void k_gru(const bf16_t* __restrict__ gi,                // (128000,192) bf16
           const float* __restrict__ h0,
           const float* __restrict__ Whh,                // (192,64)
           const float* __restrict__ bhh,
           const float* __restrict__ Wp,                 // (64,16)
           const float* __restrict__ bp,
           float* __restrict__ out) {                    // (B,N,PW)
    __shared__ float sWhh[192 * 64];
    __shared__ float sWp[64 * 16];
    __shared__ float sbhh[192];
    __shared__ float sbp[16];
    __shared__ float sh[B_ * 64];
    __shared__ float sgh[B_ * 192];
    const int tid = threadIdx.x;

    for (int i = tid; i < 192 * 64; i += 512) sWhh[i] = Whh[i];
    for (int i = tid; i < 64 * 16;  i += 512) sWp[i]  = Wp[i];
    if (tid < 192) sbhh[tid] = bhh[tid];
    if (tid < 16)  sbp[tid]  = bp[tid];
    if (tid < 512) sh[tid] = h0[tid];
    __syncthreads();

    const int gb = tid >> 6, gg = tid & 63;

    for (int step = 0; step < NT_; ++step) {
        // prefetch gate block 8 steps ahead (global_prefetch_b8)
        if (tid < 6 && step + 8 < NT_)
            __builtin_prefetch(gi + (size_t)(step + 8) * (B_ * 192) + tid * 256, 0, 0);
        // gh[b][j] = bhh[j] + h[b][:] . Whh[j][:]   (1536 dots / 512 lanes)
        #pragma unroll
        for (int p = 0; p < 3; ++p) {
            int o = tid + p * 512;
            int b = o / 192, j = o % 192;
            const float4* hr = (const float4*)(sh + b * 64);
            const float4* wr = (const float4*)(sWhh + j * 64);
            float acc = 0.f;
            #pragma unroll
            for (int k = 0; k < 16; ++k) {
                float4 hv = hr[k], wv = wr[k];
                acc += hv.x * wv.x + hv.y * wv.y + hv.z * wv.z + hv.w * wv.w;
            }
            sgh[b * 192 + j] = acc + sbhh[j];
        }
        __syncthreads();
        const bf16_t* gir = gi + ((size_t)step * B_ + gb) * 192;
        float r  = sigm((float)gir[gg]       + sgh[gb * 192 + gg]);
        float zg = sigm((float)gir[64 + gg]  + sgh[gb * 192 + 64 + gg]);
        float nn = tanhf((float)gir[128 + gg] + r * sgh[gb * 192 + 128 + gg]);
        float hn = (1.f - zg) * nn + zg * sh[gb * 64 + gg];
        __syncthreads();
        sh[gb * 64 + gg] = hn;
        __syncthreads();
        if ((step & (T_ - 1)) == T_ - 1 && tid < B_ * PW_) {
            int b = tid >> 4, p = tid & 15;
            const float* hr = sh + b * 64;
            float acc = sbp[p];
            #pragma unroll
            for (int g = 0; g < 64; ++g) acc += hr[g] * sWp[g * 16 + p];
            out[((size_t)b * N_ + (step / T_)) * PW_ + p] = acc;
        }
    }
}

// ---------------------------------------------------------------- launcher
extern "C" void kernel_launch(void* const* d_in, const int* in_sizes, int n_in,
                              void* d_out, int out_size, void* d_ws, size_t ws_size,
                              hipStream_t stream) {
    (void)in_sizes; (void)n_in; (void)out_size; (void)ws_size;
    const float* dynamic = (const float*)d_in[0];
    const float* h0      = (const float*)d_in[1];
    const int*   src     = (const int*)  d_in[2];
    const float* W1      = (const float*)d_in[3];
    const float* b1      = (const float*)d_in[4];
    const float* a1      = (const float*)d_in[5];
    const float* a1b     = (const float*)d_in[6];
    const float* W2      = (const float*)d_in[7];
    const float* b2      = (const float*)d_in[8];
    const float* a2      = (const float*)d_in[9];
    const float* a2b     = (const float*)d_in[10];
    const float* Wih     = (const float*)d_in[11];
    const float* Whh     = (const float*)d_in[12];
    const float* bih     = (const float*)d_in[13];
    const float* bhh     = (const float*)d_in[14];
    const float* Wp      = (const float*)d_in[15];
    const float* bp      = (const float*)d_in[16];
    float* out = (float*)d_out;

    char* ws = (char*)d_ws;
    size_t cur = 0;
    auto carve = [&](size_t bytes) {
        void* p = ws + cur;
        cur += (bytes + 255) & ~(size_t)255;
        return p;
    };
    bf16_t* W1b   = (bf16_t*)carve((size_t)H_ * F_ * D1_ * 2);
    bf16_t* W2b   = (bf16_t*)carve((size_t)256 * 64 * 2);
    bf16_t* WihTb = (bf16_t*)carve((size_t)64 * 192 * 2);
    float*  z1    = (float*) carve((size_t)BT_ * H_ * N_ * 64 * 4);
    float*  s1s   = (float*) carve((size_t)BT_ * H_ * N_ * 4);
    float*  s1d   = (float*) carve((size_t)BT_ * H_ * N_ * 4);
    bf16_t* h1    = (bf16_t*)carve((size_t)BT_ * N_ * 256 * 2);
    float*  z2    = (float*) carve((size_t)BT_ * N_ * 64 * 4);
    float*  s2s   = (float*) carve((size_t)BT_ * N_ * 4);
    float*  s2d   = (float*) carve((size_t)BT_ * N_ * 4);
    bf16_t* seq   = (bf16_t*)carve((size_t)M_GI * 64 * 2);
    bf16_t* gi    = (bf16_t*)carve((size_t)M_GI * 192 * 2);

    k_pack_weights<<<(61440 + 255) / 256, 256, 0, stream>>>(W1, W2, Wih, W1b, W2b, WihTb);

    k_gat1_gemm<<<dim3((NTILE_ + 3) / 4, H_, BT_), 128, 0, stream>>>(
        dynamic, W1b, b1, a1, z1, s1s, s1d);
    k_gat1_aggr<<<dim3(N_, BT_), 256, 0, stream>>>(src, z1, s1s, s1d, a1b, h1);

    k_gat2_gemm<<<dim3((NTILE_ + 3) / 4, BT_), 128, 0, stream>>>(
        h1, W2b, b2, a2, z2, s2s, s2d);
    k_gat2_aggr<<<dim3(N_, BT_), 64, 0, stream>>>(src, z2, s2s, s2d, a2b, seq);

    k_gi_gemm<<<dim3(M_GI / 64), 128, 0, stream>>>(seq, WihTb, bih, gi);

    k_gru<<<1, 512, 0, stream>>>(gi, h0, Whh, bhh, Wp, bp, out);
}
